// LinearAttention_249108103231
// MI455X (gfx1250) — matrix operand order repaired and checked by reading, not compiled, not run on hardware
//
#include <hip/hip_runtime.h>
#include <hip/hip_bf16.h>

// ---------------------------------------------------------------------------
// Linformer attention for MI455X (gfx1250): all matmuls via
// v_wmma_f32_16x16x32_bf16 (wave32, f32 accumulation), softmax in f32.
// Global->LDS staging via global_load_async_to_lds_b128 (ASYNCcnt), with
// double-buffered k-tiles in the big GEMMs (s_wait_asynccnt N pipelining).
// All WMMA fragment reads are contiguous 16B LDS vector loads.
// ---------------------------------------------------------------------------

typedef __bf16 bf16;
typedef __attribute__((ext_vector_type(16))) __bf16 bf16x16;
typedef __attribute__((ext_vector_type(8)))  __bf16 bf16x8;
typedef __attribute__((ext_vector_type(8)))  float  f32x8;
typedef __attribute__((ext_vector_type(4)))  float  f32x4;

constexpr int B_  = 4;
constexpr int S_  = 4096;
constexpr int D_  = 1024;
constexpr int H_  = 16;
constexpr int R_  = 256;
constexpr int DK_ = 64;

// ---------------------------------------------------------------------------
// CDNA5 async global->LDS copy (ISA §10.7/§15.18.3, tracked by ASYNCcnt).
// LDS byte offset = low 32 bits of the generic shared address (aperture rule).
// ---------------------------------------------------------------------------
__device__ __forceinline__ void async_copy_b128(void* lds_dst, const void* gsrc) {
    unsigned ldsoff = (unsigned)(uintptr_t)lds_dst;
    asm volatile("global_load_async_to_lds_b128 %0, %1, off"
                 :: "v"(ldsoff), "v"(gsrc)
                 : "memory");
}
// Wait until at most N async ops remain outstanding (in-order completion).
template <int N>
__device__ __forceinline__ void async_wait() {
    asm volatile("s_wait_asynccnt %0" :: "n"(N) : "memory");
}

// ---------------------------------------------------------------------------
// WMMA helpers (contiguous-k fragments, k = 16*half + j, wave32)
// ---------------------------------------------------------------------------
__device__ __forceinline__ f32x8 wmma_bf16(bf16x16 a, bf16x16 b, f32x8 c) {
    return __builtin_amdgcn_wmma_f32_16x16x32_bf16(
        false, a, false, b, (short)0, c, false, false);
}

// A fragment 16x32 bf16 from [m][k]-major storage. Per lane (half=lane>>4,
// m=lane&15): element j at k=16*half+j:
// one contiguous 32B run -> 2x ds_load_b128.
__device__ __forceinline__ bf16x16 frag_a(const bf16* base, int ld,
                                          int row0, int k0, int lane) {
    const int half = lane >> 4, m = lane & 15;
    const bf16* r = base + (size_t)(row0 + m) * ld + k0 + 16 * half;
    bf16x16 a;
#pragma unroll
    for (int j = 0; j < 16; ++j) a[j] = r[j];
    return a;
}

// B fragment 32x16 bf16 from [n][k]-major storage:
// element (k,n) at base[(n0+n)*ld + k0+k]; per lane j -> k = 16*half + j:
// one contiguous 32B run -> 2x ds_load_b128.
__device__ __forceinline__ bf16x16 frag_bT(const bf16* base, int ld,
                                           int k0, int n0, int lane) {
    const int half = lane >> 4, n = lane & 15;
    const bf16* c = base + (size_t)(n0 + n) * ld + k0 + 16 * half;
    bf16x16 b;
#pragma unroll
    for (int j = 0; j < 16; ++j) b[j] = c[j];
    return b;
}

// ---------------------------------------------------------------------------
// f32 -> bf16 conversion, 8 elems/thread/iter (b128 loads + packed cvt)
// ---------------------------------------------------------------------------
__global__ __launch_bounds__(256) void cvt_f32_bf16(const float* __restrict__ in,
                                                    bf16* __restrict__ out,
                                                    size_t n8) {
    size_t i = (size_t)blockIdx.x * blockDim.x + threadIdx.x;
    const size_t stride = (size_t)gridDim.x * blockDim.x;
    for (; i < n8; i += stride) {
        const f32x4* src = (const f32x4*)(in + i * 8);
        const f32x4 a = src[0], c = src[1];
        bf16x8 o;
#pragma unroll
        for (int u = 0; u < 4; ++u) { o[u] = (bf16)a[u]; o[4 + u] = (bf16)c[u]; }
        *(bf16x8*)(out + i * 8) = o;
    }
}

// ---------------------------------------------------------------------------
// GEMM: out = X[M,K] @ W[N,K]^T + bias[N]; 256x128x32 block tiles, 8 waves,
// each wave a 64x64 sub-tile (4x4 WMMA fragments, 128 accum VGPRs).
// Double-buffered async-to-LDS staging: issue next k-tile, wait for previous
// (s_wait_asynccnt 6), so global latency overlaps the 16 WMMAs per tile.
// ---------------------------------------------------------------------------
template <bool OUT_BF16>
__global__ __launch_bounds__(256) void gemm_xwt(const bf16* __restrict__ X,
                                                const bf16* __restrict__ W,
                                                const float* __restrict__ bias,
                                                bf16* __restrict__ outb,
                                                float* __restrict__ outf,
                                                int M, int N, int K) {
    constexpr int BM = 256, BN = 128, BK = 32;
    __shared__ bf16 As[2][BM][BK + 8];   // [m][k], 80B rows (16B-aligned chunks)
    __shared__ bf16 Bs[2][BN][BK + 8];   // [n][k] = W rows verbatim

    const int tid = threadIdx.x, lane = tid & 31, wid = tid >> 5;
    const int m0 = blockIdx.y * BM, n0 = blockIdx.x * BN;
    const int wr = (wid & 3) * 64;       // wave row offset (4 row-waves)
    const int wc = (wid >> 2) * 64;      // wave col offset (2 col-waves)

    const int arow = tid;                            // A staging: 1 row/thread
    const int brow = tid >> 1, bseg = (tid & 1) * 16;
    const bf16* srcA = X + (size_t)(m0 + arow) * K;
    const bf16* srcB = W + (size_t)(n0 + brow) * K + bseg;

    auto issue = [&](int buf, int k0) {
        async_copy_b128(&As[buf][arow][0],  srcA + k0);
        async_copy_b128(&As[buf][arow][8],  srcA + k0 + 8);
        async_copy_b128(&As[buf][arow][16], srcA + k0 + 16);
        async_copy_b128(&As[buf][arow][24], srcA + k0 + 24);
        async_copy_b128(&Bs[buf][brow][bseg],     srcB + k0);
        async_copy_b128(&Bs[buf][brow][bseg + 8], srcB + k0 + 8);
    };

    f32x8 acc[4][4] = {};
    issue(0, 0);
    int buf = 0;
    for (int k0 = 0; k0 < K; k0 += BK) {
        if (k0 + BK < K) { issue(buf ^ 1, k0 + BK); async_wait<6>(); }
        else             { async_wait<0>(); }
        __syncthreads();

        bf16x16 bfg[4];
#pragma unroll
        for (int j = 0; j < 4; ++j)
            bfg[j] = frag_bT(&Bs[buf][0][0], BK + 8, 0, wc + j * 16, lane);
#pragma unroll
        for (int i = 0; i < 4; ++i) {
            bf16x16 af = frag_a(&As[buf][0][0], BK + 8, wr + i * 16, 0, lane);
#pragma unroll
            for (int j = 0; j < 4; ++j) acc[i][j] = wmma_bf16(af, bfg[j], acc[i][j]);
        }
        __syncthreads();
        buf ^= 1;
    }

    // epilogue: D layout -> VGPR g holds row g+8*half, lane&15 holds col
    const int nlo = lane & 15, hf = lane >> 4;
#pragma unroll
    for (int i = 0; i < 4; ++i)
#pragma unroll
        for (int j = 0; j < 4; ++j) {
            const int gn = n0 + wc + j * 16 + nlo;
            const float bb = bias[gn];
#pragma unroll
            for (int g = 0; g < 8; ++g) {
                const int gm = m0 + wr + i * 16 + g + 8 * hf;
                const float v = acc[i][j][g] + bb;
                if (OUT_BF16) outb[(size_t)gm * N + gn] = (bf16)v;
                else          outf[(size_t)gm * N + gn] = v;
            }
        }
}

// ---------------------------------------------------------------------------
// kp/vp: per (b,h): kp[d,r] = sum_s kproj[b,s,h*64+d] * E[h,s,r]
// grid: x = b*H+h (64), y = r-half (2), z = 0:kp/E 1:vp/F -> 256 WGs.
// M=64(d), N=128(r), K=4096(s). Both operands transposed once at staging;
// global loads for tile i+1 are software-pipelined behind tile i's WMMAs.
// ---------------------------------------------------------------------------
__global__ __launch_bounds__(256) void kv_project(const bf16* __restrict__ kproj,
                                                  const bf16* __restrict__ vproj,
                                                  const bf16* __restrict__ Eb,
                                                  const bf16* __restrict__ Fb,
                                                  bf16* __restrict__ kpb,
                                                  bf16* __restrict__ vpb) {
    __shared__ bf16 As[DK_][32 + 8];   // [d][s]
    __shared__ bf16 Bs[128][32 + 8];   // [r][s]

    const int tid = threadIdx.x, lane = tid & 31, wid = tid >> 5;
    const int bh = blockIdx.x, b = bh / H_, h = bh - b * H_;
    const int n0 = blockIdx.y * 128;
    const bf16* X   = blockIdx.z ? vproj : kproj;
    const bf16* EFh = (blockIdx.z ? Fb : Eb) + (size_t)h * S_ * R_;
    bf16*       dst = (blockIdx.z ? vpb : kpb) + (size_t)bh * DK_ * R_;

    const int wr = (wid & 3) * 16;     // d tile
    const int wc = (wid >> 2) * 64;    // r tile within this 128-half

    const int sA = tid >> 3, dsegA = (tid & 7) * 8;
    const int rsegB = (tid & 7) * 16;

    bf16x8 va, e0, e1;
    auto gload = [&](int s0) {
        va = *(const bf16x8*)&X[((size_t)b * S_ + s0 + sA) * D_ + h * DK_ + dsegA];
        const bf16* src = &EFh[(size_t)(s0 + sA) * R_ + n0 + rsegB];
        e0 = *(const bf16x8*)src;
        e1 = *(const bf16x8*)(src + 8);
    };

    f32x8 acc[4] = {};
    gload(0);
    for (int s0 = 0; s0 < S_; s0 += 32) {
        {   // transpose scatters of the pre-fetched registers
#pragma unroll
            for (int u = 0; u < 8; ++u) As[dsegA + u][sA] = va[u];
#pragma unroll
            for (int u = 0; u < 8; ++u) {
                Bs[rsegB + u][sA]     = e0[u];
                Bs[rsegB + 8 + u][sA] = e1[u];
            }
        }
        __syncthreads();
        if (s0 + 32 < S_) gload(s0 + 32);   // overlap next tile's global loads

        bf16x16 a = frag_a(&As[0][0], 32 + 8, wr, 0, lane);
        bf16x16 bb[4];
#pragma unroll
        for (int j = 0; j < 4; ++j)
            bb[j] = frag_bT(&Bs[0][0], 32 + 8, 0, wc + j * 16, lane);
#pragma unroll
        for (int j = 0; j < 4; ++j) acc[j] = wmma_bf16(a, bb[j], acc[j]);
        __syncthreads();
    }

    const int nlo = lane & 15, hf = lane >> 4;
#pragma unroll
    for (int j = 0; j < 4; ++j)
#pragma unroll
        for (int g = 0; g < 8; ++g)
            dst[(size_t)(wr + g + 8 * hf) * R_ + n0 + wc + j * 16 + nlo] = (bf16)acc[j][g];
}

// ---------------------------------------------------------------------------
// Fused attention per (b,h,32-row S tile):
//   scores = q[32,64] @ kp[64,256] / 8   (WMMA, kp staged [r][d])
//   P = softmax_r(scores)                (f32 in LDS, bf16 P overlaid)
//   out = P[32,256] @ vp^T[256,64]       (WMMA, vp staged verbatim [d][r])
// ---------------------------------------------------------------------------
__global__ __launch_bounds__(256) void attn_fused(const bf16* __restrict__ qp,
                                                  const bf16* __restrict__ kpb,
                                                  const bf16* __restrict__ vpb,
                                                  bf16* __restrict__ opre) {
    constexpr int TS = 32;
    __shared__ bf16  Qs[TS][DK_ + 8];     // [s][d]
    __shared__ bf16  KPt[R_][DK_ + 8];    // [r][d]  (B^T for scores)
    __shared__ bf16  VPs[DK_][R_ + 8];    // [d][r]  (B^T for out)
    __shared__ float Sc[TS][R_];          // f32 scores; bf16 P overlaid per-row

    const int tid = threadIdx.x, lane = tid & 31, wid = tid >> 5;
    const int s0 = blockIdx.x * TS, h = blockIdx.y, b = blockIdx.z;
    const bf16* kb = kpb + (size_t)(b * H_ + h) * DK_ * R_;
    const bf16* vb = vpb + (size_t)(b * H_ + h) * DK_ * R_;

    {   // Qs and VPs verbatim via async-to-LDS; KPt transpose scatter overlaps
        const int i = tid >> 3, dseg = (tid & 7) * 8;
        async_copy_b128(&Qs[i][dseg],
                        &qp[((size_t)b * S_ + s0 + i) * D_ + h * DK_ + dseg]);
        const int d = tid >> 2, rseg = (tid & 3) * 64;
#pragma unroll
        for (int u = 0; u < 64; u += 8)
            async_copy_b128(&VPs[d][rseg + u], &vb[(size_t)d * R_ + rseg + u]);
#pragma unroll
        for (int u = 0; u < 64; u += 8) {
            bf16x8 t = *(const bf16x8*)&kb[(size_t)d * R_ + rseg + u];
#pragma unroll
            for (int w = 0; w < 8; ++w) KPt[rseg + u + w][d] = t[w];
        }
    }
    async_wait<0>();
    __syncthreads();

    {   // scores GEMM: M=32,N=256,K=64; per wave 16 rows x 64 cols (1x4 frags)
        const int wr = (wid & 1) * 16, wc = (wid >> 1) * 64;
        f32x8 acc[4] = {};
#pragma unroll
        for (int k0 = 0; k0 < DK_; k0 += 32) {
            bf16x16 a = frag_a(&Qs[0][0], DK_ + 8, wr, k0, lane);
            bf16x16 bb[4];
#pragma unroll
            for (int j = 0; j < 4; ++j)
                bb[j] = frag_bT(&KPt[0][0], DK_ + 8, k0, wc + j * 16, lane);
#pragma unroll
            for (int j = 0; j < 4; ++j) acc[j] = wmma_bf16(a, bb[j], acc[j]);
        }
        const int nlo = lane & 15, hf = lane >> 4;
#pragma unroll
        for (int j = 0; j < 4; ++j)
#pragma unroll
            for (int g = 0; g < 8; ++g)
                Sc[wr + g + 8 * hf][wc + j * 16 + nlo] = acc[j][g];
    }
    __syncthreads();

    // softmax over R=256: one thread per row; scale 1/sqrt(64) = 0.125
    if (tid < TS) {
        const float scale = 0.125f;
        float mx = -3.4e38f;
        for (int r = 0; r < R_; ++r) mx = fmaxf(mx, Sc[tid][r] * scale);
        float sum = 0.f;
        for (int r = 0; r < R_; ++r) {
            const float e = __expf(Sc[tid][r] * scale - mx);
            Sc[tid][r] = e;
            sum += e;
        }
        const float inv = 1.f / sum;
        bf16* Pr = reinterpret_cast<bf16*>(&Sc[tid][0]);  // in-row overlay
        for (int r = 0; r < R_; ++r)                       // write r/2 <= read r
            Pr[r] = (bf16)(Sc[tid][r] * inv);
    }
    __syncthreads();

    {   // out GEMM: M=32,N=64,K=256; per wave one 16x16 tile, 8 k-steps
        const int wr = (wid & 1) * 16, wc = (wid >> 1) * 16;
        const bf16* Pb = reinterpret_cast<const bf16*>(&Sc[0][0]);  // row stride 512
        f32x8 acc = {};
#pragma unroll
        for (int k0 = 0; k0 < R_; k0 += 32) {
            bf16x16 a  = frag_a (Pb, 512, wr, k0, lane);
            bf16x16 bb = frag_bT(&VPs[0][0], R_ + 8, k0, wc, lane);
            acc = wmma_bf16(a, bb, acc);
        }
        const int nlo = lane & 15, hf = lane >> 4;
#pragma unroll
        for (int g = 0; g < 8; ++g)
            opre[((size_t)b * S_ + s0 + wr + g + 8 * hf) * D_ + h * DK_ + wc + nlo] =
                (bf16)acc[g];
    }
}

// ---------------------------------------------------------------------------
// Host orchestration: workspace carve (~312 MB), all launches on `stream`.
// ---------------------------------------------------------------------------
extern "C" void kernel_launch(void* const* d_in, const int* in_sizes, int n_in,
                              void* d_out, int out_size, void* d_ws, size_t ws_size,
                              hipStream_t stream) {
    (void)in_sizes; (void)n_in; (void)out_size; (void)ws_size;
    const float* query = (const float*)d_in[0];
    const float* key_  = (const float*)d_in[1];
    const float* value = (const float*)d_in[2];
    const float* Wq = (const float*)d_in[3];
    const float* bq = (const float*)d_in[4];
    const float* Wk = (const float*)d_in[5];
    const float* bk = (const float*)d_in[6];
    const float* Wv = (const float*)d_in[7];
    const float* bv = (const float*)d_in[8];
    const float* Wo = (const float*)d_in[9];
    const float* bo = (const float*)d_in[10];
    const float* E  = (const float*)d_in[11];
    const float* F  = (const float*)d_in[12];

    char* p = (char*)d_ws;
    auto carve = [&](size_t elems) -> bf16* {
        bf16* r = (bf16*)p;
        p += (elems * sizeof(bf16) + 255) & ~(size_t)255;
        return r;
    };

    const size_t SD = (size_t)B_ * S_ * D_;
    const size_t WN = (size_t)D_ * D_;
    const size_t EN = (size_t)H_ * S_ * R_;
    const size_t PN = (size_t)B_ * H_ * DK_ * R_;

    bf16* xq  = carve(SD); bf16* xk  = carve(SD); bf16* xv  = carve(SD);
    bf16* wqb = carve(WN); bf16* wkb = carve(WN);
    bf16* wvb = carve(WN); bf16* wob = carve(WN);
    bf16* eb  = carve(EN); bf16* fb  = carve(EN);
    bf16* qpj = carve(SD); bf16* kpj = carve(SD); bf16* vpj = carve(SD);
    bf16* kpb = carve(PN); bf16* vpb = carve(PN);
    bf16* opre = carve(SD);

    const dim3 blk(256);
    cvt_f32_bf16<<<4096, blk, 0, stream>>>(query, xq, SD / 8);
    cvt_f32_bf16<<<4096, blk, 0, stream>>>(key_,  xk, SD / 8);
    cvt_f32_bf16<<<4096, blk, 0, stream>>>(value, xv, SD / 8);
    cvt_f32_bf16<<<512,  blk, 0, stream>>>(Wq, wqb, WN / 8);
    cvt_f32_bf16<<<512,  blk, 0, stream>>>(Wk, wkb, WN / 8);
    cvt_f32_bf16<<<512,  blk, 0, stream>>>(Wv, wvb, WN / 8);
    cvt_f32_bf16<<<512,  blk, 0, stream>>>(Wo, wob, WN / 8);
    cvt_f32_bf16<<<4096, blk, 0, stream>>>(E, eb, EN / 8);
    cvt_f32_bf16<<<4096, blk, 0, stream>>>(F, fb, EN / 8);

    const int M = B_ * S_, N = D_, K = D_;
    const dim3 gg(N / 128, M / 256);  // (8, 64) = 512 WGs
    gemm_xwt<true><<<gg, blk, 0, stream>>>(xq, wqb, bq, qpj, nullptr, M, N, K);
    gemm_xwt<true><<<gg, blk, 0, stream>>>(xk, wkb, bk, kpj, nullptr, M, N, K);
    gemm_xwt<true><<<gg, blk, 0, stream>>>(xv, wvb, bv, vpj, nullptr, M, N, K);

    kv_project<<<dim3(B_ * H_, 2, 2), blk, 0, stream>>>(kpj, vpj, eb, fb, kpb, vpb);

    attn_fused<<<dim3(S_ / 32, H_, B_), blk, 0, stream>>>(qpj, kpb, vpb, opre);

    gemm_xwt<false><<<gg, blk, 0, stream>>>(opre, wob, bo, nullptr, (float*)d_out, M, N, K);
}
